// HGTLayer_17076789969451
// MI455X (gfx1250) — compile-verified, matrix operand order checked by reading, not run on hardware
//
#include <hip/hip_runtime.h>
#include <hip/hip_bf16.h>
#include <math.h>

#define kN 50000
#define kE 800000
#define kD 128
#define kH 8
#define kHD 16
#define kNT 3
#define kET 5
#define LN_EPS 1e-5f

static_assert(kN % 16 == 0, "node tiles exact");
static_assert(kE % 16 == 0, "edge tiles exact");

typedef __attribute__((ext_vector_type(16))) __bf16 v16bf;
typedef __attribute__((ext_vector_type(8)))  __bf16 v8bf;
typedef __attribute__((ext_vector_type(8)))  float  v8f;

__device__ __forceinline__ v8f wmma_bf16(v16bf a, v16bf b, v8f c) {
  // D = A(16x32 bf16) * B(32x16 bf16) + C(16x16 f32)
  return __builtin_amdgcn_wmma_f32_16x16x32_bf16(false, a, false, b, (short)0, c, false, false);
}

// 16 contiguous bf16 (two b128) -> full v16bf (pre-swizzled B tiles, padded or full-K)
__device__ __forceinline__ v16bf ld16(const __bf16* p) {
  v8bf lo = *(const v8bf*)p;
  v8bf hi = *(const v8bf*)(p + 8);
  v16bf r;
#pragma unroll
  for (int i = 0; i < 8; ++i) { r[i] = lo[i]; r[i + 8] = hi[i]; }
  return r;
}

// A operand for one K=32 step from a row-major bf16 row: chunks at +0 and +16
__device__ __forceinline__ v16bf ldA32(const __bf16* p) {
  v8bf lo = *(const v8bf*)p;
  v8bf hi = *(const v8bf*)(p + 16);
  v16bf r;
#pragma unroll
  for (int i = 0; i < 8; ++i) { r[i] = lo[i]; r[i + 8] = hi[i]; }
  return r;
}

// order-preserving float<->uint mapping for atomicMax-based segment max
__device__ __forceinline__ unsigned enc_f32(float f) {
  unsigned u = __float_as_uint(f);
  return (u & 0x80000000u) ? ~u : (u | 0x80000000u);
}
__device__ __forceinline__ float dec_f32(unsigned k) {
  return (k & 0x80000000u) ? __uint_as_float(k & 0x7FFFFFFFu) : __uint_as_float(~k);
}

// ---------------------------------------------------------------- init / conversion
__global__ void zero_kernel(unsigned* __restrict__ p, int n) {
  int i = blockIdx.x * blockDim.x + threadIdx.x;
  if (i < n) p[i] = 0u;
}

__global__ void cvt_bf16_kernel(const float* __restrict__ src, __bf16* __restrict__ dst, int n) {
  int i = blockIdx.x * blockDim.x + threadIdx.x;
  if (i < n) dst[i] = (__bf16)src[i];
}

// x[N][128] -> xpad[N][H][32] bf16 rows with upper-K half zero
__global__ void pad_x_kernel(const float* __restrict__ src, __bf16* __restrict__ dst, int total) {
  int idx = blockIdx.x * blockDim.x + threadIdx.x;   // kN*kH*32
  if (idx >= total) return;
  int j  = idx & 31;
  int nh = idx >> 5;                  // n*kH + h
  int h  = nh & 7;
  int n  = nh >> 3;
  dst[idx] = (j < 16) ? (__bf16)src[(size_t)n * kD + h * kHD + j] : (__bf16)0.0f;
}

// batched 16x16 matrices -> 32x16 B tiles with upper-K half zero (512 elems/tile)
__global__ void swz_sq32_kernel(const float* __restrict__ src, __bf16* __restrict__ dst, int nmat) {
  int idx = blockIdx.x * blockDim.x + threadIdx.x;   // nmat*512
  if (idx >= nmat * 512) return;
  int j   = idx & 15;                 // element within lane
  int L   = (idx >> 4) & 31;          // lane
  int mat = idx >> 9;
  int kloc = (j < 8) ? (j + (L >> 4) * 8) : (16 + (j - 8) + (L >> 4) * 8);
  dst[idx] = (kloc < 16) ? (__bf16)src[mat * 256 + kloc * 16 + (L & 15)] : (__bf16)0.0f;
}

// row-major [Kdim][Ndim] -> 32x16 B tiles, per-lane 16 contiguous bf16 (512 elems/tile)
__global__ void swz32_gemm_kernel(const float* __restrict__ src, __bf16* __restrict__ dst,
                                  int Kdim, int Ndim) {
  int idx = blockIdx.x * blockDim.x + threadIdx.x;
  if (idx >= Kdim * Ndim) return;
  int j = idx & 15;
  int L = (idx >> 4) & 31;
  int tile = idx >> 9;
  int ntN = Ndim >> 4;
  int kt2 = tile / ntN;
  int ct  = tile - kt2 * ntN;
  int kloc = (j < 8) ? (j + (L >> 4) * 8) : (16 + (j - 8) + (L >> 4) * 8);
  int krow = kt2 * 32 + kloc;
  int col  = ct * 16 + (L & 15);
  dst[idx] = (__bf16)src[(size_t)krow * Ndim + col];
}

// ---------------------------------------------------------------- QKV (type-masked WMMA)
__global__ void qkv_kernel(const __bf16* __restrict__ xpad, const int* __restrict__ ntype,
                           const __bf16* __restrict__ sWQ, const __bf16* __restrict__ sWK,
                           const __bf16* __restrict__ sWV,
                           __bf16* __restrict__ Qbf, __bf16* __restrict__ Kpad,
                           __bf16* __restrict__ Vbf) {
  const int lane = threadIdx.x & 31;
  const int head = threadIdx.x >> 5;      // 8 waves: one head per wave
  const int tile = blockIdx.x;            // 16 nodes
  const int row  = lane & 15;
  const int kg   = lane >> 4;

  const int anode = tile * 16 + row;
  v16bf a = ldA32(xpad + ((size_t)anode * kH + head) * 32 + kg * 8);

  v8f c0 = {};
  const size_t o0 = (size_t)(0 * kH + head) * 512 + lane * 16;
  const size_t o1 = (size_t)(1 * kH + head) * 512 + lane * 16;
  const size_t o2 = (size_t)(2 * kH + head) * 512 + lane * 16;
  v8f dq0 = wmma_bf16(a, ld16(sWQ + o0), c0);
  v8f dq1 = wmma_bf16(a, ld16(sWQ + o1), c0);
  v8f dq2 = wmma_bf16(a, ld16(sWQ + o2), c0);
  v8f dk0 = wmma_bf16(a, ld16(sWK + o0), c0);
  v8f dk1 = wmma_bf16(a, ld16(sWK + o1), c0);
  v8f dk2 = wmma_bf16(a, ld16(sWK + o2), c0);
  v8f dv0 = wmma_bf16(a, ld16(sWV + o0), c0);
  v8f dv1 = wmma_bf16(a, ld16(sWV + o1), c0);
  v8f dv2 = wmma_bf16(a, ld16(sWV + o2), c0);

#pragma unroll
  for (int r = 0; r < 8; ++r) {
    int m = r + kg * 8;
    int n = tile * 16 + m;
    int t = ntype[n];
    float qv = (t == 0) ? dq0[r] : ((t == 1) ? dq1[r] : dq2[r]);
    float kv = (t == 0) ? dk0[r] : ((t == 1) ? dk1[r] : dk2[r]);
    float vv = (t == 0) ? dv0[r] : ((t == 1) ? dv1[r] : dv2[r]);
    size_t oc = ((size_t)n * kH + head) * kHD + row;
    Qbf[oc] = (__bf16)qv;
    Vbf[oc] = (__bf16)vv;
    Kpad[((size_t)n * kH + head) * 32 + row] = (__bf16)kv;   // upper half pre-zeroed
  }
}

// ---------------------------------------------------------------- edge scores (WMMA K-transform + seg max)
__global__ void edge_score_kernel(const __bf16* __restrict__ Qbf, const __bf16* __restrict__ Kpad,
                                  const int* __restrict__ ei, const int* __restrict__ etype,
                                  const __bf16* __restrict__ sWe, const float* __restrict__ mu,
                                  float* __restrict__ sc, unsigned* __restrict__ smax) {
  const int lane = threadIdx.x & 31;
  const int head = threadIdx.x >> 5;
  const int tile = blockIdx.x;            // 16 edges
  const int row  = lane & 15;
  const int kg   = lane >> 4;
  const int e0   = tile * 16;

  const int srcn = ei[e0 + row];
  v16bf a = ldA32(Kpad + ((size_t)srcn * kH + head) * 32 + kg * 8);

  for (int et = 0; et < kET; ++et) {
    v16bf b = ld16(sWe + (size_t)(et * kH + head) * 512 + lane * 16);
    v8f c0 = {};
    v8f d = wmma_bf16(a, b, c0);
#pragma unroll
    for (int r = 0; r < 8; ++r) {
      int m  = r + kg * 8;
      int em = e0 + m;                    // uniform within each half-wave
      bool act = (etype[em] == et);
      int dstn = 0;
      float p = 0.0f;
      if (act) {
        dstn = ei[kE + em];
        float qv = (float)Qbf[((size_t)dstn * kH + head) * kHD + row];
        p = d[r] * qv;
      }
      p += __shfl_xor(p, 1, 32);
      p += __shfl_xor(p, 2, 32);
      p += __shfl_xor(p, 4, 32);
      p += __shfl_xor(p, 8, 32);
      if (act && row == 0) {
        float s = p * 0.25f * mu[head * kET + et];   // 1/sqrt(16)
        sc[(size_t)em * kH + head] = s;
        atomicMax(&smax[dstn * kH + head], enc_f32(s));
      }
    }
  }
}

// ---------------------------------------------------------------- exp + segment-sum denominator
__global__ void softmax_norm_kernel(const int* __restrict__ ei, float* __restrict__ sc,
                                    const unsigned* __restrict__ smax, float* __restrict__ den) {
  int idx = blockIdx.x * blockDim.x + threadIdx.x;   // E*H
  if (idx >= kE * kH) return;
  int e = idx >> 3;
  int h = idx & 7;
  int dstn = ei[kE + e];
  float mx = dec_f32(smax[dstn * kH + h]);
  float ex = __expf(sc[idx] - mx);
  sc[idx] = ex;
  atomicAdd(&den[dstn * kH + h], ex);
}

// ---------------------------------------------------------------- weighted V scatter
__global__ void aggregate_kernel(const int* __restrict__ ei, const float* __restrict__ sc,
                                 const float* __restrict__ den, const __bf16* __restrict__ Vbf,
                                 float* __restrict__ agg) {
  int idx = blockIdx.x * blockDim.x + threadIdx.x;   // E*H*HD
  if (idx >= kE * kH * kHD) return;
  int f = idx & 15;
  int t = idx >> 4;
  int h = t & 7;
  int e = t >> 3;
  int srcn = ei[e];
  int dstn = ei[kE + e];
  float w = sc[(size_t)e * kH + h] / (den[dstn * kH + h] + 1e-10f);
  float v = (float)Vbf[((size_t)srcn * kH + h) * kHD + f];
  atomicAdd(&agg[((size_t)dstn * kH + h) * kHD + f], v * w);
}

// ---------------------------------------------------------------- out-proj (WMMA, K=32 steps) + residual + LN1
__global__ void outproj_ln1_kernel(const float* __restrict__ x, const __bf16* __restrict__ aggbf,
                                   const __bf16* __restrict__ sWout, const float* __restrict__ bout,
                                   const float* __restrict__ g1, const float* __restrict__ bln1,
                                   float* __restrict__ h1, __bf16* __restrict__ h1bf) {
  __shared__ float ytile[16][kD];
  __shared__ float mstat[16], vstat[16];
  const int lane = threadIdx.x & 31;
  const int wave = threadIdx.x >> 5;      // 8 col tiles
  const int tile = blockIdx.x;
  const int row  = lane & 15;
  const int kg   = lane >> 4;
  const int colbase = wave * 16;
  const int an = tile * 16 + row;

  v8f c = {};
  for (int kt2 = 0; kt2 < 4; ++kt2) {     // K = 128 in 4 full-K WMMAs
    v16bf a = ldA32(aggbf + (size_t)an * kD + kt2 * 32 + kg * 8);
    v16bf b = ld16(sWout + (size_t)(kt2 * 8 + wave) * 512 + lane * 16);
    c = wmma_bf16(a, b, c);
  }
#pragma unroll
  for (int r = 0; r < 8; ++r) {
    int m = r + kg * 8;
    int n = tile * 16 + m;
    int col = colbase + row;
    ytile[m][col] = x[(size_t)n * kD + col] + c[r] + bout[col];
  }
  __syncthreads();
  if (threadIdx.x < 16) {
    int m = threadIdx.x;
    float s = 0.0f;
    for (int j = 0; j < kD; ++j) s += ytile[m][j];
    float mean = s * (1.0f / kD);
    float v = 0.0f;
    for (int j = 0; j < kD; ++j) { float dd = ytile[m][j] - mean; v += dd * dd; }
    mstat[m] = mean;
    vstat[m] = rsqrtf(v * (1.0f / kD) + LN_EPS);
  }
  __syncthreads();
  int m  = threadIdx.x >> 4;
  int f0 = (threadIdx.x & 15) * 8;
  int n  = tile * 16 + m;
#pragma unroll
  for (int j = 0; j < 8; ++j) {
    int f = f0 + j;
    float hv = (ytile[m][f] - mstat[m]) * vstat[m] * g1[f] + bln1[f];
    h1[(size_t)n * kD + f]   = hv;
    h1bf[(size_t)n * kD + f] = (__bf16)hv;
  }
}

// ---------------------------------------------------------------- fused FFN (two K=32-step WMMA GEMMs) + LN2
__global__ void ffn_ln2_kernel(const float* __restrict__ h1, const __bf16* __restrict__ h1bf,
                               const __bf16* __restrict__ sW1, const float* __restrict__ b1,
                               const __bf16* __restrict__ sW2, const float* __restrict__ b2,
                               const float* __restrict__ g2, const float* __restrict__ bln2,
                               float* __restrict__ out) {
  __shared__ __bf16 hid[16][4 * kD];      // 16 KB : GELU(hidden) tile
  __shared__ float  ytile[16][kD];        // 8 KB
  __shared__ float  mstat[16], vstat[16];

  const int lane = threadIdx.x & 31;
  const int wave = threadIdx.x >> 5;
  const int tile = blockIdx.x;
  const int row  = lane & 15;
  const int kg   = lane >> 4;
  const int an   = tile * 16 + row;

  // GEMM1: hidden(16x512) = h1(16x128) @ W1(128x512); each wave: 4 col tiles
  for (int cc = 0; cc < 4; ++cc) {
    int ct = wave + cc * 8;                 // 0..31
    v8f c = {};
    for (int kt2 = 0; kt2 < 4; ++kt2) {
      v16bf a = ldA32(h1bf + (size_t)an * kD + kt2 * 32 + kg * 8);
      v16bf b = ld16(sW1 + (size_t)(kt2 * 32 + ct) * 512 + lane * 16);
      c = wmma_bf16(a, b, c);
    }
#pragma unroll
    for (int r = 0; r < 8; ++r) {
      int m = r + kg * 8;
      int col = ct * 16 + row;
      float v = c[r] + b1[col];
      float ge = 0.5f * v * (1.0f + erff(v * 0.70710678118654752f));  // exact GELU
      hid[m][col] = (__bf16)ge;
    }
  }
  __syncthreads();

  // GEMM2: y(16x128) = hidden(16x512) @ W2(512x128) + b2 + h1; K=512 in 16 WMMAs
  {
    v8f c = {};
    for (int kt2 = 0; kt2 < 16; ++kt2) {
      v16bf a = ldA32(&hid[row][kt2 * 32 + kg * 8]);          // 2x ds_load_b128
      v16bf b = ld16(sW2 + (size_t)(kt2 * 8 + wave) * 512 + lane * 16);
      c = wmma_bf16(a, b, c);
    }
#pragma unroll
    for (int r = 0; r < 8; ++r) {
      int m = r + kg * 8;
      int n = tile * 16 + m;
      int col = wave * 16 + row;
      ytile[m][col] = c[r] + b2[col] + h1[(size_t)n * kD + col];
    }
  }
  __syncthreads();

  if (threadIdx.x < 16) {
    int m = threadIdx.x;
    float s = 0.0f;
    for (int j = 0; j < kD; ++j) s += ytile[m][j];
    float mean = s * (1.0f / kD);
    float v = 0.0f;
    for (int j = 0; j < kD; ++j) { float dd = ytile[m][j] - mean; v += dd * dd; }
    mstat[m] = mean;
    vstat[m] = rsqrtf(v * (1.0f / kD) + LN_EPS);
  }
  __syncthreads();
  int m  = threadIdx.x >> 4;
  int f0 = (threadIdx.x & 15) * 8;
  int n  = tile * 16 + m;
#pragma unroll
  for (int j = 0; j < 8; ++j) {
    int f = f0 + j;
    out[(size_t)n * kD + f] = (ytile[m][f] - mstat[m]) * vstat[m] * g2[f] + bln2[f];
  }
}

// ---------------------------------------------------------------- launcher
extern "C" void kernel_launch(void* const* d_in, const int* in_sizes, int n_in,
                              void* d_out, int out_size, void* d_ws, size_t ws_size,
                              hipStream_t stream) {
  (void)in_sizes; (void)n_in; (void)out_size; (void)ws_size;
  const float* x     = (const float*)d_in[0];
  const int*   ei    = (const int*)d_in[1];
  const int*   etype = (const int*)d_in[2];
  const int*   ntype = (const int*)d_in[3];
  const float* WQ    = (const float*)d_in[4];
  const float* WK    = (const float*)d_in[5];
  const float* WV    = (const float*)d_in[6];
  const float* We    = (const float*)d_in[7];
  const float* mu    = (const float*)d_in[8];
  const float* Wout  = (const float*)d_in[9];
  const float* bout  = (const float*)d_in[10];
  const float* g1    = (const float*)d_in[11];
  const float* bln1  = (const float*)d_in[12];
  const float* W1    = (const float*)d_in[13];
  const float* b1    = (const float*)d_in[14];
  const float* W2    = (const float*)d_in[15];
  const float* b2    = (const float*)d_in[16];
  const float* g2    = (const float*)d_in[17];
  const float* bln2  = (const float*)d_in[18];
  float* out = (float*)d_out;

  const size_t ND = (size_t)kN * kD;   // 6.4M
  const size_t EH = (size_t)kE * kH;   // 6.4M
  const size_t NH = (size_t)kN * kH;   // 0.4M

  // fp32 region
  float* fws = (float*)d_ws;
  float* sc  = fws;                               // E*H
  unsigned* smax = (unsigned*)(sc + EH);          // N*H
  float* den = (float*)(smax + NH);               // N*H
  float* agg = den + NH;                          // N*D
  float* h1  = agg + ND;                          // N*D
  // bf16 region
  __bf16* bws   = (__bf16*)(h1 + ND);
  __bf16* xpad  = bws;            // N*H*32 = 2*ND
  __bf16* Kpad  = xpad + 2 * ND;  // N*H*32 = 2*ND
  __bf16* Qbf   = Kpad + 2 * ND;  // N*D
  __bf16* Vbf   = Qbf + ND;       // N*D
  __bf16* aggbf = Vbf + ND;       // N*D
  __bf16* h1bf  = aggbf + ND;     // N*D
  __bf16* sWQ   = h1bf + ND;      // 24*512
  __bf16* sWK   = sWQ + kNT * kH * 512;
  __bf16* sWV   = sWK + kNT * kH * 512;
  __bf16* sWe   = sWV + kNT * kH * 512;           // 40*512
  __bf16* sWout = sWe + kET * kH * 512;           // 128*128
  __bf16* sW1   = sWout + kD * kD;                // 128*512
  __bf16* sW2   = sW1 + kD * 4 * kD;              // 512*128

  const int ntiles = kN / 16;    // 3125
  const int etiles = kE / 16;    // 50000

  // zero smax + den + agg (contiguous fp32) and Kpad (bf16, 2*ND elems = ND dwords)
  int zcount = (int)(2 * NH + ND);
  zero_kernel<<<(zcount + 255) / 256, 256, 0, stream>>>(smax, zcount);
  zero_kernel<<<((int)ND + 255) / 256, 256, 0, stream>>>((unsigned*)Kpad, (int)ND);

  // one-time conversions / weight swizzles (tiny)
  pad_x_kernel<<<(kN * kH * 32 + 255) / 256, 256, 0, stream>>>(x, xpad, kN * kH * 32);
  swz_sq32_kernel<<<(kNT * kH * 512 + 255) / 256, 256, 0, stream>>>(WQ, sWQ, kNT * kH);
  swz_sq32_kernel<<<(kNT * kH * 512 + 255) / 256, 256, 0, stream>>>(WK, sWK, kNT * kH);
  swz_sq32_kernel<<<(kNT * kH * 512 + 255) / 256, 256, 0, stream>>>(WV, sWV, kNT * kH);
  swz_sq32_kernel<<<(kET * kH * 512 + 255) / 256, 256, 0, stream>>>(We, sWe, kET * kH);
  swz32_gemm_kernel<<<(kD * kD + 255) / 256, 256, 0, stream>>>(Wout, sWout, kD, kD);
  swz32_gemm_kernel<<<(kD * 4 * kD + 255) / 256, 256, 0, stream>>>(W1, sW1, kD, 4 * kD);
  swz32_gemm_kernel<<<(4 * kD * kD + 255) / 256, 256, 0, stream>>>(W2, sW2, 4 * kD, kD);

  qkv_kernel<<<ntiles, 256, 0, stream>>>(xpad, ntype, sWQ, sWK, sWV, Qbf, Kpad, Vbf);

  edge_score_kernel<<<etiles, 256, 0, stream>>>(Qbf, Kpad, ei, etype, sWe, mu, sc, smax);

  softmax_norm_kernel<<<((int)EH + 255) / 256, 256, 0, stream>>>(ei, sc, smax, den);

  aggregate_kernel<<<(kE * kH * kHD + 255) / 256, 256, 0, stream>>>(ei, sc, den, Vbf, agg);

  cvt_bf16_kernel<<<((int)ND + 255) / 256, 256, 0, stream>>>(agg, aggbf, (int)ND);

  outproj_ln1_kernel<<<ntiles, 256, 0, stream>>>(x, aggbf, sWout, bout, g1, bln1, h1, h1bf);

  ffn_ln2_kernel<<<ntiles, 256, 0, stream>>>(h1, h1bf, sW1, b1, sW2, b2, g2, bln2, out);
}
